// WaveAttentionLayer_77618648973386
// MI455X (gfx1250) — matrix operand order repaired and checked by reading, not compiled, not run on hardware
//
#include <hip/hip_runtime.h>
#include <hip/hip_bf16.h>

typedef _Float16 f16;
typedef __attribute__((ext_vector_type(16))) _Float16 v16h;
typedef __attribute__((ext_vector_type(8)))  _Float16 v8h;
typedef __attribute__((ext_vector_type(8)))  float    v8f;
typedef unsigned int u32x4 __attribute__((ext_vector_type(4)));
typedef int          i32x8 __attribute__((ext_vector_type(8)));
typedef int          i32x4 __attribute__((ext_vector_type(4)));

#define SEQ    512
#define HID    1024
#define HEADS  16
#define HD     64
#define NFREQ  33
#define KF     96
#define NKSTEP (HID / 32)

// ---- CDNA5 TDM availability / arity detection -----------------------------
#if defined(__has_builtin)
#  if __has_builtin(__builtin_amdgcn_tensor_load_to_lds) && \
      __has_builtin(__builtin_amdgcn_s_wait_tensorcnt)
#    define HAVE_TDM 1
#  else
#    define HAVE_TDM 0
#  endif
#else
#  define HAVE_TDM 0
#endif
#if defined(__has_include)
#  if __has_include(<hip/amd_detail/amd_gfx1250_TDM.h>)
#    define TDM_6ARG 1   // therock-10.0 headers -> clang-23 6-arg builtin
#  else
#    define TDM_6ARG 0   // ROCm 7.2 clang-22 5-arg builtin
#  endif
#else
#  define TDM_6ARG 0
#endif

// ---------------------------------------------------------------------------
// WMMA fragment helpers (wave32, v_wmma_f32_16x16x32_f16)
// ---------------------------------------------------------------------------
__device__ __forceinline__ v16h load_a_frag(const f16* A, int lda, int lane) {
  const int m = lane & 15, h = lane >> 4;
  const f16* p = A + m * lda + 8 * h;
  union { v16h v; v8h g[2]; } u;
  u.g[0] = *(const v8h*)(p);
  u.g[1] = *(const v8h*)(p + 16);
  return u.v;
}
__device__ __forceinline__ v16h load_b_frag(const f16* Bt, int ldb, int lane) {
  const int n = lane & 15, h = lane >> 4;
  const f16* p = Bt + n * ldb + 8 * h;
  union { v16h v; v8h g[2]; } u;
  u.g[0] = *(const v8h*)(p);
  u.g[1] = *(const v8h*)(p + 16);
  return u.v;
}
__device__ __forceinline__ v8f wmma16(v16h a, v16h b, v8f c) {
  return __builtin_amdgcn_wmma_f32_16x16x32_f16(false, a, false, b, (short)0, c,
                                                false, false);
}

// ---------------------------------------------------------------------------
// TDM: async 2-D tile DMA  global(f16, row-stride) -> LDS (packed rows).
// ---------------------------------------------------------------------------
#if HAVE_TDM
__device__ __forceinline__ void tdm_load_tile(f16* lds_dst, const f16* gsrc,
                                              int rows, int cols,
                                              int row_stride) {
  const unsigned long long ga = (unsigned long long)gsrc;
  const unsigned lds = (unsigned)(unsigned long long)lds_dst;  // LDS byte offset
  u32x4 g0;
  g0[0] = 1u;                                      // count=1 valid descriptor
  g0[1] = lds;
  g0[2] = (unsigned)(ga & 0xffffffffull);
  g0[3] = (unsigned)((ga >> 32) & 0x01ffffffull) | (2u << 30);  // type=2
  i32x8 g1;
  g1[0] = 1 << 16;                                 // data_size = 2 bytes
  g1[1] = (cols & 0xffff) << 16;                   // tensor_dim0 [63:48]
  g1[2] = (rows & 0xffff) << 16;                   // tensor_dim1 [95:80]
  g1[3] = (cols & 0xffff) << 16;                   // tile_dim0   [127:112]
  g1[4] = (rows & 0xffff);                         // tile_dim1   [143:128]
  g1[5] = row_stride;                              // dim0_stride [191:160]
  g1[6] = 0;
  g1[7] = 0;
  i32x4 z4 = {0, 0, 0, 0};
#if TDM_6ARG
  i32x8 z8 = {0, 0, 0, 0, 0, 0, 0, 0};
  __builtin_amdgcn_tensor_load_to_lds(g0, g1, z4, z4, z8, 0);
#else
  __builtin_amdgcn_tensor_load_to_lds(g0, g1, z4, z4, 0);
#endif
}
#endif

// Fallback synchronous cooperative stage (16B chunks, coalesced)
__device__ __forceinline__ void stage_tile_sync(f16* dst, const f16* src,
                                                int rows, int cols,
                                                int row_stride, int tid,
                                                int nthr) {
  const int cpr = cols >> 3;
  for (int c = tid; c < rows * cpr; c += nthr) {
    const int r = c / cpr, co = (c % cpr) * 8;
    *(v8h*)(dst + r * cols + co) = *(const v8h*)(src + r * row_stride + co);
  }
}

// ---------------------------------------------------------------------------
// 1) x f32 -> f16
// ---------------------------------------------------------------------------
__global__ void k_cvt_x(const float* __restrict__ x, f16* __restrict__ xh, int n) {
  int i = blockIdx.x * blockDim.x + threadIdx.x;
  if (i < n) xh[i] = (f16)x[i];
}

// 2) LDS-tiled transpose+convert: W[k][n] f32 -> Wt[n][k] f16
__global__ void k_cvt_wt(const float* __restrict__ W, f16* __restrict__ Wt) {
  __shared__ f16 tile[64][68];
  const int tx = threadIdx.x;                       // 256 threads
  const int tr = blockIdx.y, tc = blockIdx.x;
#pragma unroll
  for (int i = 0; i < 16; ++i) {
    const int idx = tx + i * 256, r = idx >> 6, cq = idx & 63;
    tile[cq][r] = (f16)W[(tr * 64 + r) * HID + tc * 64 + cq];
  }
  __syncthreads();
#pragma unroll
  for (int i = 0; i < 16; ++i) {
    const int idx = tx + i * 256, r = idx >> 6, cq = idx & 63;
    Wt[(tc * 64 + r) * HID + tr * 64 + cq] = tile[r][cq];
  }
}

// ---------------------------------------------------------------------------
// 3) QKV projection GEMM, WMMA + TDM double-buffered B tile in LDS.
//    grid (48, 8), block 128 (4 waves), block tile 64x64.
// ---------------------------------------------------------------------------
__global__ void k_qkv_gemm(const f16* __restrict__ Xh,
                           const f16* __restrict__ Wqt, const f16* __restrict__ Wkt,
                           const f16* __restrict__ Wvt,
                           const float* __restrict__ bq, const float* __restrict__ bk,
                           const float* __restrict__ bv,
                           float* __restrict__ Qraw, float* __restrict__ Kraw,
                           f16* __restrict__ Vt) {
  __shared__ __align__(16) f16 Bs[2][64 * 32];
  const int tid = threadIdx.x, wv = tid >> 5, lane = tid & 31;
  const int mat = blockIdx.x >> 4;
  const int xb  = blockIdx.x & 15;
  const int m0  = blockIdx.y * 64 + wv * 16;
  const f16*   Wt   = (mat == 0) ? Wqt : (mat == 1) ? Wkt : Wvt;
  const float* bias = (mat == 0) ? bq  : (mat == 1) ? bk  : bv;
  const f16* Bsrc = Wt + (xb * 64) * HID;
  const f16* Arow = Xh + m0 * HID;

  v8f c[4] = {};
  v16h a_cur = load_a_frag(Arow, HID, lane);
#if HAVE_TDM
  if (wv == 0) tdm_load_tile(&Bs[0][0], Bsrc, 64, 32, HID);
#endif
  for (int kt = 0; kt < NKSTEP; ++kt) {
    const int buf = kt & 1;
#if HAVE_TDM
    if (wv == 0) {
      if (kt + 1 < NKSTEP) {
        tdm_load_tile(&Bs[buf ^ 1][0], Bsrc + (kt + 1) * 32, 64, 32, HID);
        __builtin_amdgcn_s_wait_tensorcnt(1);   // tile kt landed; kt+1 in flight
      } else {
        __builtin_amdgcn_s_wait_tensorcnt(0);
      }
    }
    __syncthreads();
#else
    __syncthreads();
    stage_tile_sync(&Bs[buf][0], Bsrc + kt * 32, 64, 32, HID, tid, 128);
    __syncthreads();
#endif
    v16h a_nxt = a_cur;
    if (kt + 1 < NKSTEP) {
      a_nxt = load_a_frag(Arow + (kt + 1) * 32, HID, lane);
      if (kt + 2 < NKSTEP) __builtin_prefetch(Arow + (kt + 2) * 32, 0, 1);
    }
    // batch all B-fragment DS loads so dscnt waits stagger across the WMMAs
    v16h b0 = load_b_frag(&Bs[buf][0 * 16 * 32], 32, lane);
    v16h b1 = load_b_frag(&Bs[buf][1 * 16 * 32], 32, lane);
    v16h b2 = load_b_frag(&Bs[buf][2 * 16 * 32], 32, lane);
    v16h b3 = load_b_frag(&Bs[buf][3 * 16 * 32], 32, lane);
    c[0] = wmma16(a_cur, b0, c[0]);
    c[1] = wmma16(a_cur, b1, c[1]);
    c[2] = wmma16(a_cur, b2, c[2]);
    c[3] = wmma16(a_cur, b3, c[3]);
    a_cur = a_nxt;
    __syncthreads();            // all waves done with buf before DMA overwrites
  }

  const int n = lane & 15, half = lane >> 4;
#pragma unroll
  for (int j = 0; j < 4; ++j) {
    const int col  = xb * 64 + j * 16 + n;
    const float bb = bias[col];
    const int head = col >> 6, hd = col & 63;
#pragma unroll
    for (int r = 0; r < 8; ++r) {
      const int m = m0 + r + 8 * half;
      const float val = c[j][r] + bb;
      if (mat == 2)      Vt[((head << 6) + hd) * SEQ + m] = (f16)val;
      else if (mat == 0) Qraw[(head * SEQ + m) * HD + hd] = val;
      else               Kraw[(head * SEQ + m) * HD + hd] = val;
    }
  }
}

// ---------------------------------------------------------------------------
// 4) 64-pt real DFT + u = F * |F|^{-1/2} factorization (temperature in uQ)
// ---------------------------------------------------------------------------
__global__ void k_dft(const float* __restrict__ src, f16* __restrict__ u,
                      const float* __restrict__ tptr, int apply_temp) {
  __shared__ float twc[64], tws[64];
  if (threadIdx.x < 64) {
    float ang = (float)threadIdx.x * (6.28318530717958647692f / 64.0f);
    float s, cc;
    sincosf(ang, &s, &cc);
    twc[threadIdx.x] = cc;
    tws[threadIdx.x] = s;
  }
  __syncthreads();

  const int row = blockIdx.x * blockDim.x + threadIdx.x;
  const float tscale = apply_temp ? tptr[0] : 1.0f;
  float xv[HD];
  const float* p = src + row * HD;
#pragma unroll
  for (int i = 0; i < HD; ++i) xv[i] = p[i];

  f16* ur = u + row * KF;
  for (int f = 0; f < NFREQ; ++f) {
    float re = 0.f, im = 0.f;
#pragma unroll 8
    for (int nn = 0; nn < HD; ++nn) {
      const int idx = (f * nn) & 63;
      re += xv[nn] * twc[idx];
      im -= xv[nn] * tws[idx];
    }
    const float amp   = sqrtf(re * re + im * im);
    const float scale = rsqrtf(amp + 1e-20f) * tscale;
    ur[2 * f]     = (f16)(re * scale);
    ur[2 * f + 1] = (f16)(im * scale);
  }
#pragma unroll
  for (int j = 2 * NFREQ; j < KF; ++j) ur[j] = (f16)0.f;
}

// ---------------------------------------------------------------------------
// 5) scores (WMMA, K=96) -> mask+softmax in LDS -> attn x V (WMMA, K=512)
//    grid (32, 16), block 128 (4 waves)
// ---------------------------------------------------------------------------
__global__ void k_attn(const f16* __restrict__ uQ, const f16* __restrict__ uK,
                       const f16* __restrict__ Vt, const int* __restrict__ mask,
                       f16* __restrict__ ao) {
  __shared__ __align__(16) float sc[16 * SEQ];
  __shared__ __align__(16) f16   pr[16 * SEQ];
  __shared__ float red[16 * 8];
  __shared__ float rowstat[16];
  __shared__ int   mlds[SEQ];

  const int tid = threadIdx.x, wv = tid >> 5, lane = tid & 31;
  const int q0 = blockIdx.x * 16, h = blockIdx.y;
  for (int cidx = tid; cidx < SEQ; cidx += 128) mlds[cidx] = mask[cidx];

  const f16* Aq = uQ + (h * SEQ + q0) * KF;
  const v16h a0 = load_a_frag(Aq + 0,  KF, lane);
  const v16h a1 = load_a_frag(Aq + 32, KF, lane);
  const v16h a2 = load_a_frag(Aq + 64, KF, lane);
  const int n = lane & 15, half = lane >> 4;
  for (int t = 0; t < 8; ++t) {
    const int k0 = (wv * 8 + t) * 16;
    const f16* Bk = uK + (h * SEQ + k0) * KF;
    // batch the 3 global fragment loads ahead of the WMMA chain
    v16h b0 = load_b_frag(Bk + 0,  KF, lane);
    v16h b1 = load_b_frag(Bk + 32, KF, lane);
    v16h b2 = load_b_frag(Bk + 64, KF, lane);
    v8f c = {};
    c = wmma16(a0, b0, c);
    c = wmma16(a1, b1, c);
    c = wmma16(a2, b2, c);
#pragma unroll
    for (int r = 0; r < 8; ++r) sc[(r + 8 * half) * SEQ + k0 + n] = c[r];
  }
  __syncthreads();

  const int row = tid >> 3, g = tid & 7;
  float mx = -3.0e38f;
  for (int cidx = g; cidx < SEQ; cidx += 8) {
    float v = sc[row * SEQ + cidx];
    if (mlds[cidx] == 0) v = -1.0e9f;
    sc[row * SEQ + cidx] = v;
    mx = fmaxf(mx, v);
  }
  red[row * 8 + g] = mx;
  __syncthreads();
  if (g == 0) {
    float m2 = red[row * 8];
#pragma unroll
    for (int i = 1; i < 8; ++i) m2 = fmaxf(m2, red[row * 8 + i]);
    rowstat[row] = m2;
  }
  __syncthreads();
  const float rmax = rowstat[row];
  float sum = 0.f;
  for (int cidx = g; cidx < SEQ; cidx += 8) {
    const float e = __expf(sc[row * SEQ + cidx] - rmax);
    sc[row * SEQ + cidx] = e;
    sum += e;
  }
  __syncthreads();
  red[row * 8 + g] = sum;
  __syncthreads();
  if (g == 0) {
    float s2 = 0.f;
#pragma unroll
    for (int i = 0; i < 8; ++i) s2 += red[row * 8 + i];
    rowstat[row] = 1.0f / s2;
  }
  __syncthreads();
  const float inv = rowstat[row];
  for (int cidx = g; cidx < SEQ; cidx += 8)
    pr[row * SEQ + cidx] = (f16)(sc[row * SEQ + cidx] * inv);
  __syncthreads();

  // attn x V, software-pipelined (preload next a/b while current WMMA runs)
  v8f o = {};
  const f16* Bv = Vt + ((h << 6) + wv * 16) * SEQ;
  v16h a_cur = load_a_frag((const f16*)pr, SEQ, lane);
  v16h b_cur = load_b_frag(Bv, SEQ, lane);
  for (int kt = 0; kt < SEQ / 32; ++kt) {
    v16h a_nxt = a_cur, b_nxt = b_cur;
    if (kt + 1 < SEQ / 32) {
      a_nxt = load_a_frag((const f16*)pr + (kt + 1) * 32, SEQ, lane);
      b_nxt = load_b_frag(Bv + (kt + 1) * 32, SEQ, lane);
    }
    o = wmma16(a_cur, b_cur, o);
    a_cur = a_nxt;
    b_cur = b_nxt;
  }
#pragma unroll
  for (int r = 0; r < 8; ++r) {
    const int m = r + 8 * half;
    ao[(q0 + m) * HID + (h << 6) + wv * 16 + n] = (f16)o[r];
  }
}

// ---------------------------------------------------------------------------
// 6) out projection + bias + residual, WMMA + TDM-staged B tile
//    grid (16, 8), block 128
// ---------------------------------------------------------------------------
__global__ void k_oproj(const f16* __restrict__ Ah, const f16* __restrict__ Wot,
                        const float* __restrict__ bo, const float* __restrict__ xres,
                        float* __restrict__ y) {
  __shared__ __align__(16) f16 Bs[2][64 * 32];
  const int tid = threadIdx.x, wv = tid >> 5, lane = tid & 31;
  const int m0 = blockIdx.y * 64 + wv * 16;
  const int n0 = blockIdx.x * 64;
  const f16* Bsrc = Wot + n0 * HID;
  const f16* Arow = Ah + m0 * HID;

  v8f c[4] = {};
  v16h a_cur = load_a_frag(Arow, HID, lane);
#if HAVE_TDM
  if (wv == 0) tdm_load_tile(&Bs[0][0], Bsrc, 64, 32, HID);
#endif
  for (int kt = 0; kt < NKSTEP; ++kt) {
    const int buf = kt & 1;
#if HAVE_TDM
    if (wv == 0) {
      if (kt + 1 < NKSTEP) {
        tdm_load_tile(&Bs[buf ^ 1][0], Bsrc + (kt + 1) * 32, 64, 32, HID);
        __builtin_amdgcn_s_wait_tensorcnt(1);
      } else {
        __builtin_amdgcn_s_wait_tensorcnt(0);
      }
    }
    __syncthreads();
#else
    __syncthreads();
    stage_tile_sync(&Bs[buf][0], Bsrc + kt * 32, 64, 32, HID, tid, 128);
    __syncthreads();
#endif
    v16h a_nxt = a_cur;
    if (kt + 1 < NKSTEP) {
      a_nxt = load_a_frag(Arow + (kt + 1) * 32, HID, lane);
      if (kt + 2 < NKSTEP) __builtin_prefetch(Arow + (kt + 2) * 32, 0, 1);
    }
    v16h b0 = load_b_frag(&Bs[buf][0 * 16 * 32], 32, lane);
    v16h b1 = load_b_frag(&Bs[buf][1 * 16 * 32], 32, lane);
    v16h b2 = load_b_frag(&Bs[buf][2 * 16 * 32], 32, lane);
    v16h b3 = load_b_frag(&Bs[buf][3 * 16 * 32], 32, lane);
    c[0] = wmma16(a_cur, b0, c[0]);
    c[1] = wmma16(a_cur, b1, c[1]);
    c[2] = wmma16(a_cur, b2, c[2]);
    c[3] = wmma16(a_cur, b3, c[3]);
    a_cur = a_nxt;
    __syncthreads();
  }

  const int n = lane & 15, half = lane >> 4;
#pragma unroll
  for (int j = 0; j < 4; ++j) {
    const int col = n0 + j * 16 + n;
    const float bb = bo[col];
#pragma unroll
    for (int r = 0; r < 8; ++r) {
      const int m = m0 + r + 8 * half;
      y[m * HID + col] = c[j][r] + bb + xres[m * HID + col];
    }
  }
}

// ---------------------------------------------------------------------------
// 7) LayerNorm, grid 512 x 256
// ---------------------------------------------------------------------------
__global__ void k_ln(const float* __restrict__ y, const float* __restrict__ gamma,
                     const float* __restrict__ beta, float* __restrict__ out) {
  __shared__ float r1[256], r2[256];
  const int row = blockIdx.x, tid = threadIdx.x;
  float s1 = 0.f, s2 = 0.f;
  for (int cidx = tid; cidx < HID; cidx += 256) {
    const float v = y[row * HID + cidx];
    s1 += v; s2 += v * v;
  }
  r1[tid] = s1; r2[tid] = s2;
  __syncthreads();
  for (int st = 128; st > 0; st >>= 1) {
    if (tid < st) { r1[tid] += r1[tid + st]; r2[tid] += r2[tid + st]; }
    __syncthreads();
  }
  const float mu  = r1[0] * (1.0f / HID);
  const float var = r2[0] * (1.0f / HID) - mu * mu;
  const float inv = rsqrtf(var + 1e-5f);
  for (int cidx = tid; cidx < HID; cidx += 256)
    out[row * HID + cidx] = (y[row * HID + cidx] - mu) * inv * gamma[cidx] + beta[cidx];
}

// ---------------------------------------------------------------------------
extern "C" void kernel_launch(void* const* d_in, const int* in_sizes, int n_in,
                              void* d_out, int out_size, void* d_ws, size_t ws_size,
                              hipStream_t stream) {
  const float* x    = (const float*)d_in[0];
  const int*   amsk = (const int*)  d_in[1];
  const float* Wq   = (const float*)d_in[2];
  const float* bq   = (const float*)d_in[3];
  const float* Wk   = (const float*)d_in[4];
  const float* bk   = (const float*)d_in[5];
  const float* Wv   = (const float*)d_in[6];
  const float* bv   = (const float*)d_in[7];
  const float* Wo   = (const float*)d_in[8];
  const float* bo   = (const float*)d_in[9];
  const float* temp = (const float*)d_in[10];
  const float* lng  = (const float*)d_in[11];
  const float* lnb  = (const float*)d_in[12];
  float* out = (float*)d_out;

  char* ws = (char*)d_ws;
  size_t off = 0;
  auto carve = [&](size_t bytes) {
    char* p = ws + off;
    off = (off + bytes + 255) & ~(size_t)255;
    return p;
  };
  f16*   xh   = (f16*)  carve((size_t)SEQ * HID * 2);
  f16*   wqt  = (f16*)  carve((size_t)HID * HID * 2);
  f16*   wkt  = (f16*)  carve((size_t)HID * HID * 2);
  f16*   wvt  = (f16*)  carve((size_t)HID * HID * 2);
  f16*   wot  = (f16*)  carve((size_t)HID * HID * 2);
  float* Qraw = (float*)carve((size_t)HEADS * SEQ * HD * 4);
  float* Kraw = (float*)carve((size_t)HEADS * SEQ * HD * 4);
  f16*   Vt   = (f16*)  carve((size_t)HEADS * HD * SEQ * 2);
  f16*   uQ   = (f16*)  carve((size_t)HEADS * SEQ * KF * 2);
  f16*   uK   = (f16*)  carve((size_t)HEADS * SEQ * KF * 2);
  f16*   ao   = (f16*)  carve((size_t)SEQ * HID * 2);
  float* ybuf = (float*)carve((size_t)SEQ * HID * 4);

  k_cvt_x<<<(SEQ * HID) / 256, 256, 0, stream>>>(x, xh, SEQ * HID);
  k_cvt_wt<<<dim3(16, 16), 256, 0, stream>>>(Wq, wqt);
  k_cvt_wt<<<dim3(16, 16), 256, 0, stream>>>(Wk, wkt);
  k_cvt_wt<<<dim3(16, 16), 256, 0, stream>>>(Wv, wvt);
  k_cvt_wt<<<dim3(16, 16), 256, 0, stream>>>(Wo, wot);

  k_qkv_gemm<<<dim3(48, 8), 128, 0, stream>>>(xh, wqt, wkt, wvt, bq, bk, bv,
                                              Qraw, Kraw, Vt);

  k_dft<<<(HEADS * SEQ) / 128, 128, 0, stream>>>(Qraw, uQ, temp, 1);
  k_dft<<<(HEADS * SEQ) / 128, 128, 0, stream>>>(Kraw, uK, temp, 0);

  k_attn<<<dim3(SEQ / 16, HEADS), 128, 0, stream>>>(uQ, uK, Vt, amsk, ao);

  k_oproj<<<dim3(HID / 64, SEQ / 64), 128, 0, stream>>>(ao, wot, bo, x, ybuf);

  k_ln<<<SEQ, 256, 0, stream>>>(ybuf, lng, lnb, out);
  (void)in_sizes; (void)n_in; (void)out_size; (void)ws_size;
}
